// CustomGRU_28020366639565
// MI455X (gfx1250) — matrix operand order repaired and checked by reading, not compiled, not run on hardware
//
#include <hip/hip_runtime.h>
#include <hip/hip_bf16.h>

// GRU on MI455X (gfx1250): bf16 WMMA GEMMs + persistent scan kernel.
//
//   S=512, B=64, I=1024, H=1024, G=3H=3072
//
// Pipeline (all on `stream`):
//   k_bar_init   : zero per-bt grid-barrier counters (ws is poisoned, must self-init)
//   k_f2bf       : x (f32) -> xbf (bf16, row-major (S*B, I)); init_state -> hbuf[0]
//   k_pack_w     : W_ih, W_hh (f32 row-major (3H,K)) -> per-lane WMMA B-fragment order (bf16)
//   k_gi_gemm    : gi = xbf @ W_ih^T + b_ih  (LDS A tile, depth-2 pipelined B loads)
//   k_gru_scan   : persistent, 512 steps, per-bt 8-block barrier, LDS h tile,
//                  depth-4 software pipeline on the W_hh fragment stream

typedef __attribute__((ext_vector_type(16))) __bf16 v16bf;
typedef __attribute__((ext_vector_type(8)))  __bf16 v8bf;
typedef __attribute__((ext_vector_type(8)))  float  v8f;

#define S_ 512
#define B_ 64
#define I_ 1024
#define H_ 1024
#define G_ 3072
#define NSCAN_BLOCKS 32

static __device__ __forceinline__ __bf16 f2bf(float f) {
  union { float f; unsigned u; } c; c.f = f;
  unsigned u = c.u;
  unsigned r = (u + 0x7FFFu + ((u >> 16) & 1u)) >> 16;   // round-to-nearest-even
  union { unsigned short s; __bf16 b; } o; o.s = (unsigned short)r;
  return o.b;
}

static __device__ __forceinline__ float sigmoidf_(float x) {
  return 1.0f / (1.0f + __expf(-x));
}
static __device__ __forceinline__ float tanhf_(float x) {
  return 1.0f - 2.0f / (__expf(2.0f * x) + 1.0f);
}

#define WMMA_BF16(A, Bm, C) \
  __builtin_amdgcn_wmma_f32_16x16x32_bf16(false, (A), false, (Bm), (short)0, (C), false, false)

// B-fragment load: one contiguous 32B chunk per lane; eoff is a bf16-element offset
// (compile-time-constant in the hot loops so it folds into the instruction IOFFSET).
static __device__ __forceinline__ v16bf ld_b(const __bf16* p, int eoff) {
  return *(const v16bf*)(p + eoff);
}

// A-fragment (16x32 bf16, MxK) per CDNA5 ISA layout, from a row-major LDS tile.
// `ap` is pre-offset to (row = lane&15, col = (lane>>4)*8); eoff selects the k0 column.
//   lanes 0-15 : M=lane,    elems [0..7]=K k0+0..7,  [8..15]=K k0+16..23
//   lanes16-31 : M=lane-16, elems [0..7]=K k0+8..15, [8..15]=K k0+24..31
static __device__ __forceinline__ v16bf ld_a(const __bf16* ap, int eoff) {
  v8bf lo = *(const v8bf*)(ap + eoff);
  v8bf hi = *(const v8bf*)(ap + eoff + 16);
  v16bf a;
#pragma unroll
  for (int i = 0; i < 8; i++) { a[i] = lo[i]; a[i + 8] = hi[i]; }
  return a;
}

// ---------------- setup kernels ----------------

__global__ void k_bar_init(unsigned* bar) {
  if (blockIdx.x == 0 && threadIdx.x < 128) bar[threadIdx.x] = 0u;
}

__global__ void k_f2bf(const float* __restrict__ in, __bf16* __restrict__ out, int n) {
  int i = blockIdx.x * blockDim.x + threadIdx.x;
  if (i < n) out[i] = f2bf(in[i]);
}

// Pack W (f32 row-major, rows x 1024) into bf16 B-fragments.
// B-fragment (32x16 bf16, KxN): lane l, elem e -> K = k0 + (e>>3)*16 + (l>>4)*8 + (e&7), N = n0 + (l&15).
// Flat packed index i = ((ntile*32 + ktile)*32 + lane)*16 + e  (one 32B load/lane/frag).
__global__ void k_pack_w(const float* __restrict__ W, __bf16* __restrict__ out, int total) {
  int i = blockIdx.x * blockDim.x + threadIdx.x;
  if (i >= total) return;
  int e     = i & 15;
  int lane  = (i >> 4) & 31;
  int tile  = i >> 9;
  int kt    = tile & 31;          // K/32 == 32 for K==1024
  int ntile = tile >> 5;
  int grow  = ntile * 16 + (lane & 15);
  int gcol  = kt * 32 + (e >> 3) * 16 + (lane >> 4) * 8 + (e & 7);
  out[i] = f2bf(W[(size_t)grow * 1024 + gcol]);
}

// ---------------- phase 1: gi = xbf @ W_ih^T + b_ih ----------------
// M = S*B = 32768 rows, N = 3072, K = 1024.
// 8 waves/block share one M-tile (48 N-groups per M-tile, 48 % 8 == 0):
// the 16x1024 A tile is staged once in LDS. Depth-2 peeled pipeline on B.

__global__ void __launch_bounds__(256)
k_gi_gemm(const __bf16* __restrict__ xbf, const __bf16* __restrict__ wp,
          const float* __restrict__ bias, float* __restrict__ gi) {
  __shared__ __bf16 lds_a[16 * 1024];               // 32 KB
  const int NTG = G_ / 16 / 4;                      // 48 groups of 4 N-tiles
  int wave0 = blockIdx.x * 8;
  int mt    = wave0 / NTG;                          // same for all waves in block
  int wave  = wave0 + (threadIdx.x >> 5);
  int ng    = wave % NTG;
  int lane  = threadIdx.x & 31;
  int rl = lane & 15, half = lane >> 4;

  // cooperative copy: 16 rows x 1024 cols bf16 -> LDS (2048 x 16B chunks)
  {
    const v8bf* src = (const v8bf*)(xbf + (size_t)mt * 16 * I_);
    v8bf*       dst = (v8bf*)lds_a;
    for (int c = threadIdx.x; c < 2048; c += 256) dst[c] = src[c];
  }
  __syncthreads();

  const __bf16* ap = lds_a + rl * 1024 + half * 8;              // + kt*32 per k-step
  const __bf16* bp = wp + (size_t)(ng * 4) * 32 * 512 + lane * 16;
  const int JOFF = 32 * 512;                                    // N-tile stride (elems)

  v8f acc[4] = {};
  v16bf A0, A1, B0[4], B1[4];
  A0 = ld_a(ap, 0);
  A1 = ld_a(ap, 32);
#pragma unroll
  for (int j = 0; j < 4; j++) { B0[j] = ld_b(bp, j * JOFF); B1[j] = ld_b(bp, j * JOFF + 512); }

  for (int kt = 0; kt < 30; kt += 2) {
#pragma unroll
    for (int j = 0; j < 4; j++) acc[j] = WMMA_BF16(A0, B0[j], acc[j]);
    A0 = ld_a(ap, 2 * 32);
#pragma unroll
    for (int j = 0; j < 4; j++) B0[j] = ld_b(bp, j * JOFF + 2 * 512);
#pragma unroll
    for (int j = 0; j < 4; j++) acc[j] = WMMA_BF16(A1, B1[j], acc[j]);
    A1 = ld_a(ap, 3 * 32);
#pragma unroll
    for (int j = 0; j < 4; j++) B1[j] = ld_b(bp, j * JOFF + 3 * 512);
    ap += 2 * 32;
    bp += 2 * 512;
  }
#pragma unroll
  for (int j = 0; j < 4; j++) acc[j] = WMMA_BF16(A0, B0[j], acc[j]);
#pragma unroll
  for (int j = 0; j < 4; j++) acc[j] = WMMA_BF16(A1, B1[j], acc[j]);

  int m0 = mt * 16;
#pragma unroll
  for (int j = 0; j < 4; j++) {
    int n0 = (ng * 4 + j) * 16;
    float bv = bias[n0 + rl];
#pragma unroll
    for (int v = 0; v < 8; v++) {
      int row = m0 + v + 8 * half;    // C layout: VGPR v -> M = v + 8*(lane>>4), N = lane&15
      gi[(size_t)row * G_ + n0 + rl] = acc[j][v] + bv;
    }
  }
}

// ---------------- phase 2: persistent GRU scan ----------------
// 32 blocks x 256 threads = 256 waves = (B/16)=4 b-tiles x (H/16)=64 h-tiles.
// All 8 waves in a block share one b-tile -> h A-tile staged in LDS each step.
// The 4 b-tiles are independent recurrences: each syncs only its own 8 blocks.
// Depth-4 peeled pipeline: a stage's reload has 9 WMMAs in front of its wait.

__global__ void __launch_bounds__(256)
k_gru_scan(const float* __restrict__ gi, const __bf16* __restrict__ whh_p,
           const float* __restrict__ bias_hh, const float* __restrict__ init_state,
           __bf16* __restrict__ hbuf, float* __restrict__ out, unsigned* __restrict__ bar) {
  __shared__ __bf16 lds_h[16 * 1024];               // 32 KB: rows bt*16..+15 of h
  int wave = blockIdx.x * 8 + (threadIdx.x >> 5);   // 0..255
  int lane = threadIdx.x & 31;
  int bt = wave >> 6;            // 0..3  (constant per block)
  int ht = wave & 63;            // 0..63
  int rl = lane & 15, half = lane >> 4;
  int hcol = ht * 16 + rl;
  unsigned* mybar = bar + bt * 32;                  // per-bt barrier, own cache line

  float bhr = bias_hh[hcol];
  float bhz = bias_hh[H_ + hcol];
  float bhn = bias_hh[2 * H_ + hcol];

  float hold[8];
#pragma unroll
  for (int v = 0; v < 8; v++) {
    int b = bt * 16 + v + 8 * half;
    hold[v] = init_state[(size_t)b * H_ + hcol];
  }

  const __bf16* ap_base = lds_h + rl * 1024 + half * 8;
  const __bf16* bp_base = whh_p + (size_t)ht * 32 * 512 + lane * 16;
  const int ZOFF = 64 * 32 * 512;                   // gate z fragment offset (elems, 2MB)
  const int NOFF = 128 * 32 * 512;                  // gate n fragment offset (elems, 4MB)

  for (int s = 0; s < S_; s++) {
    const __bf16* hb = hbuf + (size_t)(s & 1) * (B_ * H_);
    __bf16*       hn = hbuf + (size_t)((s + 1) & 1) * (B_ * H_);

    // stage this block's 16 h-rows into LDS (2048 x 16B chunks)
    {
      const v8bf* src = (const v8bf*)(hb + (size_t)bt * 16 * H_);
      v8bf*       dst = (v8bf*)lds_h;
      for (int c = threadIdx.x; c < 2048; c += 256) dst[c] = src[c];
    }
    __syncthreads();

    // issue gi loads early: independent of the WMMA chain
    float gir[8], giz[8], gin[8];
#pragma unroll
    for (int v = 0; v < 8; v++) {
      int b = bt * 16 + v + 8 * half;
      const float* gp = gi + ((size_t)s * B_ + b) * G_;
      gir[v] = gp[hcol];
      giz[v] = gp[H_ + hcol];
      gin[v] = gp[2 * H_ + hcol];
    }

    v8f ar = {}, az = {}, an = {};
    const __bf16* ap = ap_base;
    const __bf16* bp = bp_base;

    v16bf A[4], R[4], Z[4], N[4];
#pragma unroll
    for (int q = 0; q < 4; q++) {
      A[q] = ld_a(ap, q * 32);
      R[q] = ld_b(bp, q * 512);
      Z[q] = ld_b(bp, ZOFF + q * 512);
      N[q] = ld_b(bp, NOFF + q * 512);
    }
    for (int kt = 0; kt < 28; kt += 4) {
#pragma unroll
      for (int q = 0; q < 4; q++) {
        ar = WMMA_BF16(A[q], R[q], ar);
        az = WMMA_BF16(A[q], Z[q], az);
        an = WMMA_BF16(A[q], N[q], an);
        A[q] = ld_a(ap, (4 + q) * 32);
        R[q] = ld_b(bp, (4 + q) * 512);
        Z[q] = ld_b(bp, ZOFF + (4 + q) * 512);
        N[q] = ld_b(bp, NOFF + (4 + q) * 512);
      }
      ap += 4 * 32;
      bp += 4 * 512;
    }
#pragma unroll
    for (int q = 0; q < 4; q++) {
      ar = WMMA_BF16(A[q], R[q], ar);
      az = WMMA_BF16(A[q], Z[q], az);
      an = WMMA_BF16(A[q], N[q], an);
    }

#pragma unroll
    for (int v = 0; v < 8; v++) {
      int b = bt * 16 + v + 8 * half;
      float r  = sigmoidf_(gir[v] + ar[v] + bhr);
      float z  = sigmoidf_(giz[v] + az[v] + bhz);
      float n  = tanhf_   (gin[v] + r * (an[v] + bhn));
      float hv = n + z * (hold[v] - n);
      hold[v] = hv;
      out[(size_t)s * (B_ * H_) + (size_t)b * H_ + hcol] = hv;
      hn[(size_t)b * H_ + hcol] = f2bf(hv);
      if (s == S_ - 1)
        out[(size_t)S_ * (B_ * H_) + (size_t)b * H_ + hcol] = hv;  // h_last
    }

    // barrier over the 8 blocks sharing this bt (independent recurrences per bt)
    __threadfence();
    __syncthreads();
    if (threadIdx.x == 0) {
      atomicAdd(mybar, 1u);
      unsigned target = (unsigned)(s + 1) * 8u;
      volatile unsigned* vb = mybar;
      while (*vb < target) __builtin_amdgcn_s_sleep(2);
    }
    __syncthreads();
    __threadfence();
  }
}

// ---------------- host-side launch ----------------

extern "C" void kernel_launch(void* const* d_in, const int* in_sizes, int n_in,
                              void* d_out, int out_size, void* d_ws, size_t ws_size,
                              hipStream_t stream) {
  const float* x          = (const float*)d_in[0];   // (S,B,I)
  const float* init_state = (const float*)d_in[1];   // (B,H)
  const float* w_ih       = (const float*)d_in[2];   // (3H,I)
  const float* w_hh       = (const float*)d_in[3];   // (3H,H)
  const float* b_ih       = (const float*)d_in[4];   // (3H)
  const float* b_hh       = (const float*)d_in[5];   // (3H)
  float* out = (float*)d_out;                        // (S,B,H) ++ (B,H)

  char* ws = (char*)d_ws;
  // workspace layout (all offsets 256B aligned)
  float*  gi     = (float*) (ws + 0);                           // 402,653,184 B
  __bf16* xbf    = (__bf16*)(ws + 402653184ull);                //  67,108,864 B
  __bf16* wih_p  = (__bf16*)(ws + 469762048ull);                //   6,291,456 B
  __bf16* whh_p  = (__bf16*)(ws + 476053504ull);                //   6,291,456 B
  __bf16* hbuf   = (__bf16*)(ws + 482344960ull);                //     262,144 B (double-buffered)
  unsigned* bar  = (unsigned*)(ws + 482607104ull);              //         512 B (4 counters)

  // barrier counters must be zeroed every call (ws not re-poisoned between replays)
  k_bar_init<<<1, 128, 0, stream>>>(bar);

  // x -> bf16
  {
    int n = S_ * B_ * I_;
    k_f2bf<<<(n + 255) / 256, 256, 0, stream>>>(x, xbf, n);
  }
  // init_state -> hbuf[0]
  {
    int n = B_ * H_;
    k_f2bf<<<(n + 255) / 256, 256, 0, stream>>>(init_state, hbuf, n);
  }
  // pack weights into WMMA B-fragment order
  {
    int total = G_ * I_;   // 3,145,728
    k_pack_w<<<(total + 255) / 256, 256, 0, stream>>>(w_ih, wih_p, total);
    k_pack_w<<<(total + 255) / 256, 256, 0, stream>>>(w_hh, whh_p, total);
  }
  // gi GEMM: (S*B/16) * (G/64) = 2048 * 48 = 98304 waves -> 12288 blocks of 8 waves
  {
    int waves  = (S_ * B_ / 16) * (G_ / 64);
    int blocks = waves / 8;
    k_gi_gemm<<<blocks, 256, 0, stream>>>(xbf, wih_p, b_ih, gi);
  }
  // persistent scan
  k_gru_scan<<<NSCAN_BLOCKS, 256, 0, stream>>>(gi, whh_p, b_hh, init_state, hbuf, out, bar);
}
